// MALAAttention_43834436223024
// MI455X (gfx1250) — matrix operand order repaired and checked by reading, not compiled, not run on hardware
//
#include <hip/hip_runtime.h>
#include <hip/hip_bf16.h>

typedef __attribute__((ext_vector_type(16))) __bf16 v16bf;
typedef __attribute__((ext_vector_type(8)))  float  f32x8;
typedef __attribute__((ext_vector_type(8)))  __bf16 bf16x8;
typedef __attribute__((ext_vector_type(4)))  int    v4i;

union frag_u { uint4 q[2]; v16bf v; };

// ---------------------------------------------------------------------------
// Async global->LDS copy (gfx1250 GLOBAL_LOAD_ASYNC_TO_LDS_B128, ASYNCcnt)
// ---------------------------------------------------------------------------
#if defined(__gfx1250__) && __has_builtin(__builtin_amdgcn_global_load_async_to_lds_b128)
#define HAVE_ASYNC_LDS 1
#endif

#ifdef HAVE_ASYNC_LDS
typedef __attribute__((address_space(1))) v4i* g4_t;   // global int4*
typedef __attribute__((address_space(3))) v4i* l4_t;   // LDS int4*
__device__ __forceinline__ void cp_b128(void* lds, const void* g) {
  __builtin_amdgcn_global_load_async_to_lds_b128((g4_t)g, (l4_t)lds, 0, 0);
}
__device__ __forceinline__ void cp_wait() {
#if __has_builtin(__builtin_amdgcn_s_wait_asynccnt)
  __builtin_amdgcn_s_wait_asynccnt(0);
#else
  asm volatile("s_wait_asynccnt 0" ::: "memory");
#endif
}
#else
__device__ __forceinline__ void cp_b128(void* lds, const void* g) {
  *(uint4*)lds = *(const uint4*)g;
}
__device__ __forceinline__ void cp_wait() {}
#endif

// ---------------------------------------------------------------------------
// WMMA helpers (gfx1250, wave32).  D = A(16x32 bf16) x B(32x16 bf16) + C(f32)
// ---------------------------------------------------------------------------
__device__ __forceinline__ f32x8 wmma_bf16(v16bf a, v16bf b, f32x8 c) {
  return __builtin_amdgcn_wmma_f32_16x16x32_bf16(
      false, a, false, b, (short)0, c, false, false);
}

__device__ __forceinline__ int a_kpair(int v, int half) {
  return 2 * v + half * 16;
}

// A fragment from LDS tile [m][k]: two contiguous 16B runs -> 2x ds_load_b128
__device__ __forceinline__ v16bf load_a_frag_vec(const __bf16* As, int ld, int mbase,
                                                 int kofs, int lane) {
  int m = mbase + (lane & 15);
  int half = lane >> 4;
  const __bf16* base = As + m * ld + kofs;
  frag_u u;
  u.q[0] = *(const uint4*)(base + half * 16);
  u.q[1] = *(const uint4*)(base + half * 16 + 8);
  return u.v;
}

// B fragment from LDS tile [n][k]: one contiguous 16-half run -> 2x b128
__device__ __forceinline__ v16bf load_b_frag_vec(const __bf16* Bs, int ld, int nbase,
                                                 int kofs, int lane) {
  int n = nbase + (lane & 15);
  int koff = (lane >> 4) * 16 + kofs;
  const __bf16* base = Bs + n * ld + koff;
  frag_u u;
  u.q[0] = *(const uint4*)(base);
  u.q[1] = *(const uint4*)(base + 8);
  return u.v;
}

// Scalar variants for the (tiny) kv kernel: staged [n][m] / [n][e]
__device__ __forceinline__ v16bf load_a_frag_t(const __bf16* Ks, int ld, int mbase,
                                               int kofs, int lane) {
  v16bf a;
  int m = mbase + (lane & 15);
  int half = lane >> 4;
#pragma unroll
  for (int v = 0; v < 8; ++v) {
    int kk = a_kpair(v, half) + kofs;
    a[2 * v]     = Ks[kk * ld + m];
    a[2 * v + 1] = Ks[(kk + 1) * ld + m];
  }
  return a;
}
__device__ __forceinline__ v16bf load_b_frag_kn(const __bf16* Bs, int ld, int nbase,
                                                int kofs, int lane) {
  v16bf b;
  int n = nbase + (lane & 15);
  int koff = (lane >> 4) * 16 + kofs;
#pragma unroll
  for (int v = 0; v < 8; ++v) {
    b[2 * v]     = Bs[(koff + 2 * v) * ld + n];
    b[2 * v + 1] = Bs[(koff + 2 * v + 1) * ld + n];
  }
  return b;
}

__device__ __forceinline__ f32x8 zero8() {
  f32x8 z = {0.f, 0.f, 0.f, 0.f, 0.f, 0.f, 0.f, 0.f};
  return z;
}
__device__ __forceinline__ bf16x8 cvt8(float4 a, float4 b) {
  bf16x8 v = {(__bf16)a.x, (__bf16)a.y, (__bf16)a.z, (__bf16)a.w,
              (__bf16)b.x, (__bf16)b.y, (__bf16)b.z, (__bf16)b.w};
  return v;
}

// ---------------------------------------------------------------------------
// f32 -> bf16 bulk convert (one-shot; makes GEMM staging a pure byte copy)
// ---------------------------------------------------------------------------
__global__ void __launch_bounds__(256)
cvt_bf16_kernel(const float* __restrict__ src, __bf16* __restrict__ dst, int n) {
  int i = (blockIdx.x * 256 + threadIdx.x) * 8;
  if (i < n) {
    float4 a = *(const float4*)(src + i);
    float4 b = *(const float4*)(src + i + 4);
    *(bf16x8*)(dst + i) = cvt8(a, b);
  }
}

// ---------------------------------------------------------------------------
// GEMM:  C[M,N] = A[M,K] @ W[N,K]^T + bias  (A, W pre-converted bf16)
// 128x128 tile / block, 8 waves (4m x 2n), each wave 2x4 16x16 subtiles.
// Double-buffered LDS with async global->LDS staging.
// ---------------------------------------------------------------------------
template <bool ELU>
__global__ void __launch_bounds__(256)
gemm_bias_kernel(const __bf16* __restrict__ A, const __bf16* __restrict__ W,
                 const float* __restrict__ bias, float* __restrict__ Cout,
                 int M, int N, int K) {
  __shared__ __attribute__((aligned(16))) __bf16 As[2][128 * 40];  // [m][k]
  __shared__ __attribute__((aligned(16))) __bf16 Bs[2][128 * 40];  // [n][k]
  const int tid  = threadIdx.x;
  const int lane = tid & 31;
  const int wave = tid >> 5;
  const int wm   = wave & 3;     // m offset = wm*32
  const int wn   = wave >> 2;    // n offset = wn*64
  const int m0 = blockIdx.x * 128;
  const int n0 = blockIdx.y * 128;

  f32x8 acc[2][4];
#pragma unroll
  for (int i = 0; i < 2; ++i)
#pragma unroll
    for (int j = 0; j < 4; ++j) acc[i][j] = zero8();

  // stage one 128x32 bf16 tile of A and of W: 512 b128 chunks each, 2/thread
  auto stage = [&](int k0, int buf) {
#pragma unroll
    for (int c2 = 0; c2 < 2; ++c2) {
      int chunk = c2 * 256 + tid;
      int r = chunk >> 2, col = (chunk & 3) * 8;
      cp_b128(&As[buf][r * 40 + col], A + (size_t)(m0 + r) * K + k0 + col);
      cp_b128(&Bs[buf][r * 40 + col], W + (size_t)(n0 + r) * K + k0 + col);
    }
  };

  const int nk = K >> 5;
  stage(0, 0);
  cp_wait();
  __syncthreads();

  for (int kt = 0; kt < nk; ++kt) {
    const int buf = kt & 1;
    if (kt + 1 < nk) stage((kt + 1) * 32, buf ^ 1);   // DMA during the WMMAs

    v16bf a[2], b[4];
#pragma unroll
    for (int sm = 0; sm < 2; ++sm)
      a[sm] = load_a_frag_vec(As[buf], 40, wm * 32 + sm * 16, 0, lane);
#pragma unroll
    for (int sn = 0; sn < 4; ++sn)
      b[sn] = load_b_frag_vec(Bs[buf], 40, wn * 64 + sn * 16, 0, lane);
#pragma unroll
    for (int sm = 0; sm < 2; ++sm)
#pragma unroll
      for (int sn = 0; sn < 4; ++sn) acc[sm][sn] = wmma_bf16(a[sm], b[sn], acc[sm][sn]);

    if (kt + 1 < nk) {
      cp_wait();
      __syncthreads();
    }
  }

  const int colb   = lane & 15;
  const int rowoff = (lane >> 4) * 8;
#pragma unroll
  for (int sn = 0; sn < 4; ++sn) {
    int nn = n0 + wn * 64 + sn * 16 + colb;
    float bv = bias[nn];
#pragma unroll
    for (int sm = 0; sm < 2; ++sm) {
      int mbase = m0 + wm * 32 + sm * 16 + rowoff;
#pragma unroll
      for (int r = 0; r < 8; ++r) {
        float v = acc[sm][sn][r] + bv;
        if (ELU) v = (v > 0.f) ? (v + 1.f) : __expf(v);  // elu(x)+1
        Cout[(size_t)(mbase + r) * N + nn] = v;
      }
    }
  }
}

// ---------------------------------------------------------------------------
// Per-(b,channel) means of K (elu'd) and V over n.
// ---------------------------------------------------------------------------
__global__ void __launch_bounds__(256)
mean_kernel(const float* __restrict__ K, const float* __restrict__ V,
            float* __restrict__ kmean, float* __restrict__ vmean) {
  const int c = blockIdx.x, b = blockIdx.y, which = blockIdx.z;
  const float* src = which ? V : K;
  const int tid = threadIdx.x;
  float s = 0.f;
#pragma unroll
  for (int i = 0; i < 16; ++i) {
    int n = tid + i * 256;
    s += src[(size_t)(b * 4096 + n) * 768 + c];
  }
  __shared__ float red[256];
  red[tid] = s;
  __syncthreads();
  for (int off = 128; off; off >>= 1) {
    if (tid < off) red[tid] += red[tid + off];
    __syncthreads();
  }
  if (tid == 0) (which ? vmean : kmean)[b * 768 + c] = red[0] * (1.f / 4096.f);
}

// ---------------------------------------------------------------------------
// Fused: z = scale * <q, kmean>;  2D-RoPE theta-shift on q,k;  bf16 casts.
// ---------------------------------------------------------------------------
__global__ void __launch_bounds__(256)
rope_z_kernel(const float* __restrict__ Q, const float* __restrict__ K,
              const float* __restrict__ V, const float* __restrict__ kmean,
              __bf16* __restrict__ Qh, __bf16* __restrict__ Kh,
              __bf16* __restrict__ Vh, float* __restrict__ z) {
  const int tid  = threadIdx.x;
  const int rloc = tid >> 6;
  const int d    = tid & 63;
  const int row  = blockIdx.x * 4 + rloc;
  const int b = row >> 12, n = row & 4095;
  const int pi = n >> 6, pj = n & 63;
  const float pos = (d < 32) ? (float)pi : (float)pj;
  const int t2 = ((d < 32) ? d : d - 32) >> 1;
  const float ang = __powf(10000.f, -(float)t2 * (1.f / 15.f));
  const float sv = __sinf(pos * ang), cv = __cosf(pos * ang);
  const float scale = 0.125f;
  __shared__ float red[256];

  for (int h = 0; h < 12; ++h) {
    const int off = row * 768 + h * 64 + d;
    float q = Q[off], k = K[off], v = V[off];
    red[tid] = q * kmean[b * 768 + h * 64 + d];
    __syncthreads();
    if (d == 0) {
      float zs = 0.f;
      for (int t = 0; t < 64; ++t) zs += red[rloc * 64 + t];
      z[(b * 12 + h) * 4096 + n] = zs * scale;
    }
    float qp = __shfl_xor(q, 1, 32);
    float kp = __shfl_xor(k, 1, 32);
    float qrot = (d & 1) ? qp : -qp;
    float krot = (d & 1) ? kp : -kp;
    Qh[off] = (__bf16)(q * cv + qrot * sv);
    Kh[off] = (__bf16)(k * cv + krot * sv);
    Vh[off] = (__bf16)v;
    __syncthreads();
  }
}

__global__ void zero_f32(float* p, int n) {
  int i = blockIdx.x * 256 + threadIdx.x;
  if (i < n) p[i] = 0.f;
}

// ---------------------------------------------------------------------------
// kv[b,h] (64x64) += k_hat^T @ v_hat over a 512-row chunk of n (WMMA).
// ---------------------------------------------------------------------------
__global__ void __launch_bounds__(256)
kv_accum_kernel(const __bf16* __restrict__ Kh, const __bf16* __restrict__ Vh,
                float* __restrict__ kvacc) {
  const int bh = blockIdx.x;
  const int b = bh / 12, h = bh % 12;
  const int chunk = blockIdx.y;
  __shared__ __attribute__((aligned(16))) __bf16 Ks[32 * 72];  // [n][d]
  __shared__ __attribute__((aligned(16))) __bf16 Vs[32 * 72];  // [n][e]
  const int tid = threadIdx.x, lane = tid & 31, wave = tid >> 5;
  const int wm = wave >> 1;
  const int wn = wave & 1;
  f32x8 acc[2] = {zero8(), zero8()};

  const int r = tid >> 3, col = (tid & 7) * 8;   // 256 b128 chunks per matrix
  for (int nb = chunk * 512; nb < chunk * 512 + 512; nb += 32) {
    size_t g = (size_t)(b * 4096 + nb + r) * 768 + h * 64 + col;
    cp_b128(&Ks[r * 72 + col], Kh + g);
    cp_b128(&Vs[r * 72 + col], Vh + g);
    cp_wait();
    __syncthreads();
    v16bf a  = load_a_frag_t(Ks, 72, wm * 16, 0, lane);
    v16bf b0 = load_b_frag_kn(Vs, 72, wn * 32, 0, lane);
    v16bf b1 = load_b_frag_kn(Vs, 72, wn * 32 + 16, 0, lane);
    acc[0] = wmma_bf16(a, b0, acc[0]);
    acc[1] = wmma_bf16(a, b1, acc[1]);
    __syncthreads();
  }
  const int colb = lane & 15, rowoff = (lane >> 4) * 8;
#pragma unroll
  for (int sn = 0; sn < 2; ++sn)
#pragma unroll
    for (int rr = 0; rr < 8; ++rr) {
      int mm = wm * 16 + rowoff + rr;
      int nn = wn * 32 + sn * 16 + colb;
      atomicAdd(&kvacc[bh * 4096 + mm * 64 + nn], acc[sn][rr]);
    }
}

// ---------------------------------------------------------------------------
// res = (q_hat @ kv) * (1 + 1/(z+1e-6)) - z * vmean      -> bf16
// ---------------------------------------------------------------------------
__global__ void __launch_bounds__(256)
res_kernel(const __bf16* __restrict__ Qh, const float* __restrict__ kv,
           const float* __restrict__ z, const float* __restrict__ vmean,
           __bf16* __restrict__ res, float kvscale) {
  const int mt = blockIdx.x, h = blockIdx.y, b = blockIdx.z;
  const int bh = b * 12 + h;
  __shared__ __attribute__((aligned(16))) __bf16 Qs[128 * 72];  // [m][k]
  __shared__ __attribute__((aligned(16))) __bf16 KVs[64 * 72];  // [e][k] (transposed)
  const int tid = threadIdx.x, lane = tid & 31, wave = tid >> 5;
  const int wm = wave & 3;
  const int wn = wave >> 2;
  const int row0 = mt * 128;

  // async-stage the first 64 rows (64x64) of the q_hat tile (512 b128 chunks, 2/thread)
#pragma unroll
  for (int c2 = 0; c2 < 2; ++c2) {
    int chunk = c2 * 256 + tid;
    int r = chunk >> 3, col = (chunk & 7) * 8;
    cp_b128(&Qs[r * 72 + col],
            Qh + (size_t)(b * 4096 + row0 + r) * 768 + h * 64 + col);
  }
  // stage kv transposed [e][k], scaled + bf16 (small, L2-hot)
#pragma unroll
  for (int it = 0; it < 2; ++it) {
    int chunk = it * 256 + tid;          // 512 chunks of 8 k's
    int e = chunk >> 3;
    int kk0 = (chunk & 7) * 8;
    bf16x8 v;
#pragma unroll
    for (int j = 0; j < 8; ++j) v[j] = (__bf16)(kv[bh * 4096 + (kk0 + j) * 64 + e] * kvscale);
    *(bf16x8*)(&KVs[e * 72 + kk0]) = v;
  }
  cp_wait();
  __syncthreads();

  f32x8 acc[2][2];
#pragma unroll
  for (int i = 0; i < 2; ++i)
#pragma unroll
    for (int j = 0; j < 2; ++j) acc[i][j] = zero8();

#pragma unroll
  for (int ks = 0; ks < 2; ++ks) {
    v16bf a[2], bf[2];
#pragma unroll
    for (int sm = 0; sm < 2; ++sm)
      a[sm] = load_a_frag_vec(Qs, 72, wm * 32 + sm * 16, ks * 32, lane);
#pragma unroll
    for (int sn = 0; sn < 2; ++sn)
      bf[sn] = load_b_frag_vec(KVs, 72, wn * 32 + sn * 16, ks * 32, lane);
#pragma unroll
    for (int sm = 0; sm < 2; ++sm)
#pragma unroll
      for (int sn = 0; sn < 2; ++sn) acc[sm][sn] = wmma_bf16(a[sm], bf[sn], acc[sm][sn]);
  }

  const int colb = lane & 15, rowoff = (lane >> 4) * 8;
#pragma unroll
  for (int sm = 0; sm < 2; ++sm)
#pragma unroll
    for (int sn = 0; sn < 2; ++sn) {
      int e = wn * 32 + sn * 16 + colb;
      float vm = vmean[bh * 64 + e];
#pragma unroll
      for (int r = 0; r < 8; ++r) {
        int rl = wm * 32 + sm * 16 + rowoff + r;
        float zv = z[bh * 4096 + row0 + rl];
        float val = acc[sm][sn][r] * (1.f + 1.f / (zv + 1e-6f)) - zv * vm;
        res[(size_t)(b * 4096 + row0 + rl) * 768 + h * 64 + e] = (__bf16)val;
      }
    }
}

// ---------------------------------------------------------------------------
extern "C" void kernel_launch(void* const* d_in, const int* in_sizes, int n_in,
                              void* d_out, int out_size, void* d_ws, size_t ws_size,
                              hipStream_t stream) {
  (void)in_sizes; (void)n_in; (void)out_size; (void)ws_size;
  const float* query = (const float*)d_in[0];
  const float* key   = (const float*)d_in[1];
  const float* value = (const float*)d_in[2];
  const float* Wq = (const float*)d_in[3];
  const float* bq = (const float*)d_in[4];
  const float* Wk = (const float*)d_in[5];
  const float* bk = (const float*)d_in[6];
  const float* Wv = (const float*)d_in[7];
  const float* bv = (const float*)d_in[8];
  const float* Wo = (const float*)d_in[9];
  const float* bo = (const float*)d_in[10];

  const int B = 4, N = 4096, C = 768, NH = 12, D = 64;
  const int M = B * N;
  const int MC = M * C, CC = C * C;

  char* w = (char*)d_ws;
  size_t off = 0;
  auto alloc = [&](size_t bytes) -> void* {
    void* p = w + off;
    off = (off + bytes + 255) & ~(size_t)255;
    return p;
  };
  float*  Q     = (float*)alloc((size_t)MC * 4);
  float*  K     = (float*)alloc((size_t)MC * 4);
  float*  V     = (float*)alloc((size_t)MC * 4);
  __bf16* Qh    = (__bf16*)alloc((size_t)MC * 2);
  __bf16* Kh    = (__bf16*)alloc((size_t)MC * 2);
  __bf16* Vh    = (__bf16*)alloc((size_t)MC * 2);
  __bf16* resb  = (__bf16*)alloc((size_t)MC * 2);
  __bf16* xq    = (__bf16*)alloc((size_t)MC * 2);   // bf16 copies of inputs
  __bf16* xk    = (__bf16*)alloc((size_t)MC * 2);
  __bf16* xv    = (__bf16*)alloc((size_t)MC * 2);
  __bf16* wqb   = (__bf16*)alloc((size_t)CC * 2);   // bf16 copies of weights
  __bf16* wkb   = (__bf16*)alloc((size_t)CC * 2);
  __bf16* wvb   = (__bf16*)alloc((size_t)CC * 2);
  __bf16* wob   = (__bf16*)alloc((size_t)CC * 2);
  float*  kmean = (float*)alloc((size_t)B * C * 4);
  float*  vmean = (float*)alloc((size_t)B * C * 4);
  float*  zbuf  = (float*)alloc((size_t)B * NH * N * 4);
  float*  kvacc = (float*)alloc((size_t)B * NH * D * D * 4);

  // 0) one-shot bf16 conversions (makes GEMM staging async-copy-only)
  cvt_bf16_kernel<<<dim3(MC / 2048), 256, 0, stream>>>(query, xq, MC);
  cvt_bf16_kernel<<<dim3(MC / 2048), 256, 0, stream>>>(key,   xk, MC);
  cvt_bf16_kernel<<<dim3(MC / 2048), 256, 0, stream>>>(value, xv, MC);
  cvt_bf16_kernel<<<dim3(CC / 2048), 256, 0, stream>>>(Wq, wqb, CC);
  cvt_bf16_kernel<<<dim3(CC / 2048), 256, 0, stream>>>(Wk, wkb, CC);
  cvt_bf16_kernel<<<dim3(CC / 2048), 256, 0, stream>>>(Wv, wvb, CC);
  cvt_bf16_kernel<<<dim3(CC / 2048), 256, 0, stream>>>(Wo, wob, CC);

  dim3 gproj(M / 128, C / 128);
  // 1) projections (elu+1 fused into q,k)
  gemm_bias_kernel<true ><<<gproj, 256, 0, stream>>>(xq, wqb, bq, Q, M, C, C);
  gemm_bias_kernel<true ><<<gproj, 256, 0, stream>>>(xk, wkb, bk, K, M, C, C);
  gemm_bias_kernel<false><<<gproj, 256, 0, stream>>>(xv, wvb, bv, V, M, C, C);
  // 2) k/v means
  mean_kernel<<<dim3(C, B, 2), 256, 0, stream>>>(K, V, kmean, vmean);
  // 3) z + RoPE + bf16 casts
  rope_z_kernel<<<dim3(M / 4), 256, 0, stream>>>(Q, K, V, kmean, Qh, Kh, Vh, zbuf);
  // 4) kv = k^T v
  zero_f32<<<dim3((B * NH * D * D + 255) / 256), 256, 0, stream>>>(kvacc, B * NH * D * D);
  kv_accum_kernel<<<dim3(B * NH, 8), 256, 0, stream>>>(Kh, Vh, kvacc);
  // 5) res = q@kv * (1+1/(z+eps)) - z*vmean
  res_kernel<<<dim3(N / 128, NH, B), 256, 0, stream>>>(Qh, kvacc, zbuf, vmean, resb,
                                                       0.125f / 4096.f);
  // 6) output projection
  gemm_bias_kernel<false><<<gproj, 256, 0, stream>>>(resb, wob, bo, (float*)d_out, M, C, C);
}
